// VectorBasedCrossAttention_54271206752777
// MI455X (gfx1250) — matrix operand order repaired and checked by reading, not compiled, not run on hardware
//
#include <hip/hip_runtime.h>
#include <hip/hip_bf16.h>
#include <math.h>

// ---------------- problem constants ----------------
#define B_  2
#define QL_ 2048
#define KL_ 4096
#define H_  1024
#define NH_ 16
#define HD_ 64
#define SCALE_ 0.125f
#define LN_EPS_ 1e-5f

typedef __attribute__((ext_vector_type(16))) _Float16 v16h;
typedef __attribute__((ext_vector_type(8)))  _Float16 v8h;
typedef __attribute__((ext_vector_type(8)))  float    v8f;
typedef int v4i_ __attribute__((__vector_size__(16)));  // matches builtin param type

union V16    { v16h v; v8h h[2]; _Float16 e[16]; };
union Pack16 { uint4 u[2]; _Float16 e[16]; };

__device__ __forceinline__ v8f wmma16(v16h a, v16h b, v8f c) {
  // (neg_a, A, neg_b, B, c_mod, C, reuse_a, reuse_b)
  return __builtin_amdgcn_wmma_f32_16x16x32_f16(false, a, false, b, (short)0, c,
                                                false, false);
}

// ---------------- async global -> LDS (CDNA5), guarded ----------------
#if defined(__gfx1250__) && __has_builtin(__builtin_amdgcn_global_load_async_to_lds_b128)
#define HAVE_ASYNC 1
#endif

__device__ __forceinline__ void async_cp16(const void* g, void* l) {
#if defined(HAVE_ASYNC)
  __builtin_amdgcn_global_load_async_to_lds_b128((v4i_*)g, (v4i_*)l, 0, 0);
#else
  *(uint4*)l = *(const uint4*)g;
#endif
}
__device__ __forceinline__ void async_wait() {
#if defined(HAVE_ASYNC)
#if __has_builtin(__builtin_amdgcn_s_wait_asynccnt)
  __builtin_amdgcn_s_wait_asynccnt(0);
#else
  asm volatile("s_wait_asynccnt 0x0" ::: "memory");
#endif
#endif
}

// ---------------- weight convert + transpose: Wt[n][k] = (f16)W[k][n] ----------------
__global__ __launch_bounds__(256) void cvt_w_t_kernel(
    const float* __restrict__ W, _Float16* __restrict__ Wt) {
  __shared__ _Float16 t[32][33];
  const int k0 = blockIdx.x * 32, n0 = blockIdx.y * 32;
  const int tk = threadIdx.x >> 3, tn4 = (threadIdx.x & 7) * 4;
  const float4 v = *(const float4*)(W + (size_t)(k0 + tk) * H_ + n0 + tn4);
  t[tn4 + 0][tk] = (_Float16)v.x;
  t[tn4 + 1][tk] = (_Float16)v.y;
  t[tn4 + 2][tk] = (_Float16)v.z;
  t[tn4 + 3][tk] = (_Float16)v.w;
  __syncthreads();
  const int rn = threadIdx.x >> 3, ck = (threadIdx.x & 7) * 4;
  _Float16* o = Wt + (size_t)(n0 + rn) * H_ + k0 + ck;
#pragma unroll
  for (int i = 0; i < 4; ++i) o[i] = t[rn][ck + i];
}

// ---------------- tiled WMMA GEMM: Y = A(MxK) * B(KxN) + bias ----------------
// B given TRANSPOSED (Bt[n][k]); K=N=1024; tile 128x128, K-step 32, double-buffered.
enum { OUTM_F16 = 0, OUTM_F32 = 1, OUTM_VT = 2 };

template <typename AT, int OUTM>
__global__ __launch_bounds__(256) void gemm_kernel(
    const AT* __restrict__ A, const _Float16* __restrict__ Bt,
    const float* __restrict__ bias, _Float16* __restrict__ Yh,
    float* __restrict__ Yf, int M) {
  const int K = H_, N = H_;
  __shared__ __align__(16) _Float16 Asl[2][128 * 40];  // padded stride 40 halves
  __shared__ __align__(16) _Float16 Btl[2][128 * 40];  // Bt rows [n][k]

  const int tid  = threadIdx.x;
  const int lane = tid & 31, wave = tid >> 5;
  const int hi   = lane >> 4, lo = lane & 15;
  const int m0 = blockIdx.x * 128;
  const int n0 = blockIdx.y * 128;
  const int wm = wave >> 1;  // 0..3 -> 32-row slab
  const int wn = wave & 1;   // 0..1 -> 64-col slab
  const int r2 = tid >> 1, c2 = (tid & 1) * 16;

  auto stage = [&](int buf, int k0) {
    if constexpr (sizeof(AT) == 2) {
      const _Float16* ga = (const _Float16*)A + (size_t)(m0 + r2) * K + k0 + c2;
      async_cp16(ga,     &Asl[buf][r2 * 40 + c2]);
      async_cp16(ga + 8, &Asl[buf][r2 * 40 + c2 + 8]);
    } else {
      const AT* ga = A + (size_t)(m0 + r2) * K + k0 + c2;
      Pack16 t;
#pragma unroll
      for (int i = 0; i < 16; ++i) t.e[i] = (_Float16)(float)ga[i];
      uint4* s = (uint4*)&Asl[buf][r2 * 40 + c2];
      s[0] = t.u[0]; s[1] = t.u[1];
    }
    const _Float16* gb = Bt + (size_t)(n0 + r2) * K + k0 + c2;
    async_cp16(gb,     &Btl[buf][r2 * 40 + c2]);
    async_cp16(gb + 8, &Btl[buf][r2 * 40 + c2 + 8]);
  };

  v8f acc[2][4];
#pragma unroll
  for (int i = 0; i < 2; ++i)
#pragma unroll
    for (int j = 0; j < 4; ++j) acc[i][j] = (v8f)0.0f;

  const int nk = K / 32;
  stage(0, 0);
  for (int it = 0; it < nk; ++it) {
    async_wait();
    __syncthreads();
    if (it + 1 < nk) stage((it + 1) & 1, (it + 1) * 32);

    const _Float16* As = Asl[it & 1];
    const _Float16* Bs = Btl[it & 1];
    V16 af[2], bf[4];
#pragma unroll
    for (int i = 0; i < 2; ++i) {
      const int rowa = wm * 32 + i * 16 + lo;
      af[i].h[0] = *(const v8h*)(As + rowa * 40 + hi * 8);
      af[i].h[1] = *(const v8h*)(As + rowa * 40 + 16 + hi * 8);
    }
#pragma unroll
    for (int j = 0; j < 4; ++j) {
      const int nrow = wn * 64 + j * 16 + lo;
      bf[j].h[0] = *(const v8h*)(Bs + nrow * 40 + hi * 8);
      bf[j].h[1] = *(const v8h*)(Bs + nrow * 40 + 16 + hi * 8);
    }
#pragma unroll
    for (int i = 0; i < 2; ++i)
#pragma unroll
      for (int j = 0; j < 4; ++j)
        acc[i][j] = wmma16(af[i].v, bf[j].v, acc[i][j]);
  }

  // epilogue: C layout — c[e] = C[8*hi + e][lo] within each 16x16 tile
#pragma unroll
  for (int i = 0; i < 2; ++i)
#pragma unroll
    for (int j = 0; j < 4; ++j) {
      const int n = n0 + wn * 64 + j * 16 + lo;
      const float bvv = bias[n];
#pragma unroll
      for (int e = 0; e < 8; ++e) {
        const int m = m0 + wm * 32 + i * 16 + 8 * hi + e;
        const float v = acc[i][j][e] + bvv;
        if constexpr (OUTM == OUTM_F16) {
          Yh[(size_t)m * N + n] = (_Float16)v;
        } else if constexpr (OUTM == OUTM_F32) {
          Yf[(size_t)m * N + n] = v;
        } else {  // OUTM_VT: Vt[((b*NH+head)*HD + d) * KL + key]
          const int bb = m >> 12, key = m & (KL_ - 1);
          const int head = n >> 6, d = n & 63;
          Yh[((size_t)(bb * NH_ + head) * HD_ + d) * KL_ + key] = (_Float16)v;
        }
      }
    }
}

// ---------------- flash attention ----------------
// grid: (QL/128, NH, B); block 256 (8 waves), each wave owns 16 q rows.
// V given pre-transposed per (b,head): Vt[d][key], so all staging is b128.
__global__ __launch_bounds__(256) void attn_kernel(
    const _Float16* __restrict__ Qh, const _Float16* __restrict__ Kh,
    const _Float16* __restrict__ Vt, const int* __restrict__ qmask,
    const int* __restrict__ astart_p, _Float16* __restrict__ Ctxh) {
  __shared__ __align__(16) _Float16 Kl[2][64 * 72];   // [key][d]
  __shared__ __align__(16) _Float16 Vl[2][64 * 72];   // [d][key]
  __shared__ __align__(16) _Float16 Pl[8 * 16 * 72];  // per-wave P relayout

  const int tid  = threadIdx.x;
  const int lane = tid & 31, wave = tid >> 5;
  const int hi   = lane >> 4, lo = lane & 15;
  const int b    = blockIdx.z;
  const int head = blockIdx.y;
  const int qb   = blockIdx.x * 128;

  int astart = *astart_p;
  if (astart < 0) astart = 0;
  if (astart > KL_) astart = KL_;
  const int kk0 = (astart >> 6) << 6;
  const int nchunk = (KL_ - kk0) >> 6;

  const int r4 = tid >> 2, c4 = (tid & 3) * 16;
  const _Float16* Kbase = Kh + ((size_t)b * KL_) * H_ + head * HD_;
  const _Float16* Vbase = Vt + (size_t)(b * NH_ + head) * HD_ * KL_;

  auto stage = [&](int buf, int kk) {
    const _Float16* gk = Kbase + (size_t)(kk + r4) * H_ + c4;      // r4=key, c4=d
    async_cp16(gk,     &Kl[buf][r4 * 72 + c4]);
    async_cp16(gk + 8, &Kl[buf][r4 * 72 + c4 + 8]);
    const _Float16* gv = Vbase + (size_t)r4 * KL_ + kk + c4;       // r4=d, c4=key
    async_cp16(gv,     &Vl[buf][r4 * 72 + c4]);
    async_cp16(gv + 8, &Vl[buf][r4 * 72 + c4 + 8]);
  };

  // resident Q fragments (A layout over d): 16 rows x 64 d
  const _Float16* Qbase = Qh + ((size_t)b * QL_) * H_ + head * HD_;
  const int qrow = qb + wave * 16 + lo;  // A-frag row = lane%16
  V16 aq[2];
#pragma unroll
  for (int ks = 0; ks < 2; ++ks) {
    aq[ks].h[0] = *(const v8h*)(Qbase + (size_t)qrow * H_ + ks * 32 + hi * 8);
    aq[ks].h[1] = *(const v8h*)(Qbase + (size_t)qrow * H_ + ks * 32 + 16 + hi * 8);
  }

  float mrun[8], lrun[8];
  v8f o[4];
#pragma unroll
  for (int e = 0; e < 8; ++e) { mrun[e] = -INFINITY; lrun[e] = 0.0f; }
#pragma unroll
  for (int dt = 0; dt < 4; ++dt) o[dt] = (v8f)0.0f;

  _Float16* pw = Pl + wave * (16 * 72);

  if (nchunk > 0) stage(0, kk0);
  for (int c = 0; c < nchunk; ++c) {
    const int kk = kk0 + c * 64;
    async_wait();
    __syncthreads();
    if (c + 1 < nchunk) stage((c + 1) & 1, kk + 64);
    const _Float16* Ks = Kl[c & 1];
    const _Float16* Vs = Vl[c & 1];

    // S = Q @ K^T  (4 key-tiles of 16, 2 k-steps of 32 over d)
    v8f s[4];
#pragma unroll
    for (int nt = 0; nt < 4; ++nt) s[nt] = (v8f)0.0f;
#pragma unroll
    for (int nt = 0; nt < 4; ++nt) {
#pragma unroll
      for (int ks = 0; ks < 2; ++ks) {
        V16 bk;
        const _Float16* kp = Ks + (nt * 16 + lo) * 72 + ks * 32 + hi * 8;
        bk.h[0] = *(const v8h*)(kp);
        bk.h[1] = *(const v8h*)(kp + 16);
        s[nt] = wmma16(aq[ks].v, bk.v, s[nt]);
      }
    }

    // scale + key mask + row max
    float mnew[8];
#pragma unroll
    for (int e = 0; e < 8; ++e) mnew[e] = mrun[e];
#pragma unroll
    for (int nt = 0; nt < 4; ++nt) {
      const int key = kk + nt * 16 + lo;
      const bool kvalid = (key >= astart);
#pragma unroll
      for (int e = 0; e < 8; ++e) {
        float v = s[nt][e] * SCALE_;
        v = kvalid ? v : -INFINITY;
        s[nt][e] = v;
        mnew[e] = fmaxf(mnew[e], v);
      }
    }
#pragma unroll
    for (int msk = 1; msk < 16; msk <<= 1)
#pragma unroll
      for (int e = 0; e < 8; ++e)
        mnew[e] = fmaxf(mnew[e], __shfl_xor(mnew[e], msk, 32));

    // rescale running state, exponentiate
    float alpha[8];
#pragma unroll
    for (int e = 0; e < 8; ++e) {
      alpha[e] = __expf(mrun[e] - mnew[e]);
      mrun[e] = mnew[e];
      lrun[e] *= alpha[e];
    }
#pragma unroll
    for (int dt = 0; dt < 4; ++dt)
#pragma unroll
      for (int e = 0; e < 8; ++e) o[dt][e] *= alpha[e];

    float lsum[8];
#pragma unroll
    for (int e = 0; e < 8; ++e) lsum[e] = 0.0f;
#pragma unroll
    for (int nt = 0; nt < 4; ++nt)
#pragma unroll
      for (int e = 0; e < 8; ++e) {
        const float p = __expf(s[nt][e] - mnew[e]);
        s[nt][e] = p;
        lsum[e] += p;
      }
#pragma unroll
    for (int msk = 1; msk < 16; msk <<= 1)
#pragma unroll
      for (int e = 0; e < 8; ++e) lsum[e] += __shfl_xor(lsum[e], msk, 32);
#pragma unroll
    for (int e = 0; e < 8; ++e) lrun[e] += lsum[e];

    // P -> LDS (C layout) -> A layout; wave-private region, DS in-order
#pragma unroll
    for (int nt = 0; nt < 4; ++nt)
#pragma unroll
      for (int e = 0; e < 8; ++e)
        pw[(8 * hi + e) * 72 + nt * 16 + lo] = (_Float16)s[nt][e];

#pragma unroll
    for (int ks = 0; ks < 2; ++ks) {
      V16 pa;
      pa.h[0] = *(const v8h*)(pw + lo * 72 + ks * 32 + hi * 8);
      pa.h[1] = *(const v8h*)(pw + lo * 72 + ks * 32 + 16 + hi * 8);
#pragma unroll
      for (int dt = 0; dt < 4; ++dt) {
        V16 bv;
        const _Float16* vp = Vs + (dt * 16 + lo) * 72 + ks * 32 + hi * 8;
        bv.h[0] = *(const v8h*)(vp);
        bv.h[1] = *(const v8h*)(vp + 16);
        o[dt] = wmma16(pa.v, bv.v, o[dt]);
      }
    }
  }

  // normalize + write ctx (f16 for the Wo GEMM)
  const size_t obase = ((size_t)b * QL_) * H_ + head * HD_;
#pragma unroll
  for (int e = 0; e < 8; ++e) {
    const int qr = qb + wave * 16 + 8 * hi + e;
    const int valid = qmask[b * QL_ + qr];
    const float inv = (valid && lrun[e] > 0.0f) ? 1.0f / lrun[e] : 0.0f;
#pragma unroll
    for (int dt = 0; dt < 4; ++dt)
      Ctxh[obase + (size_t)qr * H_ + dt * 16 + lo] = (_Float16)(o[dt][e] * inv);
  }
}

// ---------------- residual + layernorm ----------------
__global__ __launch_bounds__(256) void ln_kernel(
    const float* __restrict__ x, const float* __restrict__ pre,
    const float* __restrict__ w, const float* __restrict__ bb,
    float* __restrict__ out) {
  const int row = blockIdx.x, tid = threadIdx.x;
  const float4* xp = (const float4*)(x + (size_t)row * H_);
  const float4* pp = (const float4*)(pre + (size_t)row * H_);
  float4 v = xp[tid], p = pp[tid];
  v.x += p.x; v.y += p.y; v.z += p.z; v.w += p.w;
  float s  = v.x + v.y + v.z + v.w;
  float s2 = v.x * v.x + v.y * v.y + v.z * v.z + v.w * v.w;
#pragma unroll
  for (int m = 1; m < 32; m <<= 1) {
    s  += __shfl_xor(s,  m, 32);
    s2 += __shfl_xor(s2, m, 32);
  }
  __shared__ float rs[8], rs2[8];
  const int wave = tid >> 5, lane = tid & 31;
  if (lane == 0) { rs[wave] = s; rs2[wave] = s2; }
  __syncthreads();
  float ts = 0.0f, ts2 = 0.0f;
#pragma unroll
  for (int i = 0; i < 8; ++i) { ts += rs[i]; ts2 += rs2[i]; }
  const float mu  = ts / (float)H_;
  const float var = ts2 / (float)H_ - mu * mu;
  const float inv = rsqrtf(var + LN_EPS_);
  const float4 wv = ((const float4*)w)[tid];
  const float4 bv = ((const float4*)bb)[tid];
  float4 o;
  o.x = (v.x - mu) * inv * wv.x + bv.x;
  o.y = (v.y - mu) * inv * wv.y + bv.y;
  o.z = (v.z - mu) * inv * wv.z + bv.z;
  o.w = (v.w - mu) * inv * wv.w + bv.w;
  ((float4*)(out + (size_t)row * H_))[tid] = o;
}

// ---------------- host launch ----------------
extern "C" void kernel_launch(void* const* d_in, const int* in_sizes, int n_in,
                              void* d_out, int out_size, void* d_ws,
                              size_t ws_size, hipStream_t stream) {
  const float* q   = (const float*)d_in[0];
  const float* kn  = (const float*)d_in[1];
  const int*   am  = (const int*)d_in[2];
  const int*   ast = (const int*)d_in[3];
  const float* Wq  = (const float*)d_in[4];
  const float* bq  = (const float*)d_in[5];
  const float* Wk  = (const float*)d_in[6];
  const float* bk  = (const float*)d_in[7];
  const float* Wv  = (const float*)d_in[8];
  const float* bv  = (const float*)d_in[9];
  const float* Wo  = (const float*)d_in[10];
  const float* bo  = (const float*)d_in[11];
  const float* lnw = (const float*)d_in[12];
  const float* lnb = (const float*)d_in[13];
  float* out = (float*)d_out;

  // workspace layout (all region sizes 256B aligned)
  char* ws = (char*)d_ws;
  size_t off = 0;
  auto take = [&](size_t bytes) { void* p = ws + off; off += (bytes + 255) & ~(size_t)255; return p; };
  _Float16* Qh   = (_Float16*)take((size_t)B_ * QL_ * H_ * 2);   // row-major
  _Float16* Kh   = (_Float16*)take((size_t)B_ * KL_ * H_ * 2);   // row-major
  _Float16* Vth  = (_Float16*)take((size_t)B_ * KL_ * H_ * 2);   // [(b*NH+h)*HD+d][KL]
  _Float16* Wqt  = (_Float16*)take((size_t)H_ * H_ * 2);         // transposed weights
  _Float16* Wkt  = (_Float16*)take((size_t)H_ * H_ * 2);
  _Float16* Wvt  = (_Float16*)take((size_t)H_ * H_ * 2);
  _Float16* Wot  = (_Float16*)take((size_t)H_ * H_ * 2);
  _Float16* Ctxh = (_Float16*)take((size_t)B_ * QL_ * H_ * 2);
  float*    Pre  = (float*)take((size_t)B_ * QL_ * H_ * 4);
  (void)ws_size; (void)in_sizes; (void)n_in; (void)out_size;

  // 1) convert + transpose weights to f16
  dim3 gw(H_ / 32, H_ / 32);
  cvt_w_t_kernel<<<gw, 256, 0, stream>>>(Wq, Wqt);
  cvt_w_t_kernel<<<gw, 256, 0, stream>>>(Wk, Wkt);
  cvt_w_t_kernel<<<gw, 256, 0, stream>>>(Wv, Wvt);
  cvt_w_t_kernel<<<gw, 256, 0, stream>>>(Wo, Wot);

  // 2) projections (A = f32 activations converted while staging to LDS)
  dim3 gq(B_ * QL_ / 128, H_ / 128);
  dim3 gk(B_ * KL_ / 128, H_ / 128);
  gemm_kernel<float, OUTM_F16><<<gq, 256, 0, stream>>>(q,  Wqt, bq, Qh,  nullptr, B_ * QL_);
  gemm_kernel<float, OUTM_F16><<<gk, 256, 0, stream>>>(kn, Wkt, bk, Kh,  nullptr, B_ * KL_);
  gemm_kernel<float, OUTM_VT ><<<gk, 256, 0, stream>>>(kn, Wvt, bv, Vth, nullptr, B_ * KL_);

  // 3) flash attention
  attn_kernel<<<dim3(QL_ / 128, NH_, B_), 256, 0, stream>>>(Qh, Kh, Vth, am, ast, Ctxh);

  // 4) output projection (f32 out, bias added; A is f16 -> async staging)
  gemm_kernel<_Float16, OUTM_F32><<<gq, 256, 0, stream>>>(Ctxh, Wot, bo, nullptr, Pre, B_ * QL_);

  // 5) residual + layernorm
  ln_kernel<<<B_ * QL_, 256, 0, stream>>>(q, Pre, lnw, lnb, out);
}
